// DGNN_81303730913787
// MI455X (gfx1250) — compile-verified, matrix operand order checked by reading, not run on hardware
//
#include <hip/hip_runtime.h>

// ---------------------------------------------------------------------------
// Types
// ---------------------------------------------------------------------------
typedef __bf16 bf16;
typedef __attribute__((ext_vector_type(16))) __bf16 v16bf;
typedef __attribute__((ext_vector_type(8)))  float  v8f;

#define NN_  6144
#define HH_  128

__device__ inline v8f wmma_bf16(v16bf a, v16bf b, v8f c) {
  return __builtin_amdgcn_wmma_f32_16x16x32_bf16(
      /*neg_a=*/false, a, /*neg_b=*/false, b,
      /*c_mod=*/(short)0, c, /*reuse_a=*/false, /*reuse_b=*/false);
}

// Branch-free tanh: (e^{2x}-1)/(e^{2x}+1).
// __expf -> v_exp_f32, __builtin_amdgcn_rcpf -> v_rcp_f32 (both TRANS ops
// that co-execute with WMMA; no IEEE-div fixup chains, no branches).
__device__ inline float fast_tanh(float x) {
  float xc = fminf(fmaxf(x, -15.f), 15.f);
  float e = __expf(2.f * xc);
  return (e - 1.f) * __builtin_amdgcn_rcpf(e + 1.f);
}

// A-fragment (16x32 bf16, row-major source [rows x ld]):
// lane L (L&15) -> row, (L>>4) -> K half (16 contiguous bf16 = 32B load)
__device__ inline v16bf load_frag(const bf16* __restrict__ base, int row,
                                  int ld, int kbase, int lane) {
  const bf16* p = base + (size_t)(row + (lane & 15)) * (size_t)ld +
                  (size_t)(kbase + ((lane >> 4) << 4));
  return *(const v16bf*)p;
}

// ---------------------------------------------------------------------------
// Generic NT GEMM: C[MxNc] = A[MxK] * B[NcxK]^T   (A,B bf16 row-major)
// block = 128 threads (4 waves), 64x128 block tile, wave tile 32x64
// (2 A-frags + 4 B-frags -> 8 WMMAs per K-chunk)
// ---------------------------------------------------------------------------
template <bool BF16OUT>
__global__ __launch_bounds__(128) void k_gemm_nt(
    const bf16* __restrict__ A, const bf16* __restrict__ B,
    void* __restrict__ Cout, int M, int Nc, int K, int lda, int ldb, int ldc) {
  int lane = threadIdx.x & 31;
  int wave = threadIdx.x >> 5;
  int mbase = blockIdx.y * 64 + (wave & 1) * 32;
  int nbase = blockIdx.x * 128 + (wave >> 1) * 64;
  if (mbase >= M || nbase >= Nc) return;

  v8f acc[2][4] = {};
  for (int k0 = 0; k0 < K; k0 += 32) {
    v16bf a0 = load_frag(A, mbase,      lda, k0, lane);
    v16bf a1 = load_frag(A, mbase + 16, lda, k0, lane);
    v16bf b0 = load_frag(B, nbase,      ldb, k0, lane);
    v16bf b1 = load_frag(B, nbase + 16, ldb, k0, lane);
    v16bf b2 = load_frag(B, nbase + 32, ldb, k0, lane);
    v16bf b3 = load_frag(B, nbase + 48, ldb, k0, lane);
    acc[0][0] = wmma_bf16(a0, b0, acc[0][0]);
    acc[0][1] = wmma_bf16(a0, b1, acc[0][1]);
    acc[0][2] = wmma_bf16(a0, b2, acc[0][2]);
    acc[0][3] = wmma_bf16(a0, b3, acc[0][3]);
    acc[1][0] = wmma_bf16(a1, b0, acc[1][0]);
    acc[1][1] = wmma_bf16(a1, b1, acc[1][1]);
    acc[1][2] = wmma_bf16(a1, b2, acc[1][2]);
    acc[1][3] = wmma_bf16(a1, b3, acc[1][3]);
  }
  int mofs = (lane >> 4) * 8;
  int ncol = lane & 15;
#pragma unroll
  for (int i = 0; i < 2; ++i)
#pragma unroll
    for (int j = 0; j < 4; ++j)
#pragma unroll
      for (int r = 0; r < 8; ++r) {
        size_t m = (size_t)(mbase + i * 16 + mofs + r);
        size_t n = (size_t)(nbase + j * 16 + ncol);
        float v = acc[i][j][r];
        if (BF16OUT) ((bf16*)Cout)[m * (size_t)ldc + n] = (bf16)v;
        else         ((float*)Cout)[m * (size_t)ldc + n] = v;
      }
}

// ---------------------------------------------------------------------------
// Fused attention block (no softmax; score = tanh(q k^T), att = score v)
// grid = (N/64, HEADS), block = 128 (4 waves); wave handles a 16-row strip.
// Scores never touch HBM: WMMA -> tanh -> bf16 -> LDS -> A-frag -> WMMA.
// ---------------------------------------------------------------------------
__global__ __launch_bounds__(128) void k_att(
    const bf16* __restrict__ qB, const bf16* __restrict__ kB,
    const bf16* __restrict__ vT,  // [H x N] (transposed v)
    bf16* __restrict__ attB, int n) {
  int lane = threadIdx.x & 31;
  int wave = threadIdx.x >> 5;
  int doff = blockIdx.y * 64;               // head offset in feature dim
  int m0 = blockIdx.x * 64 + wave * 16;     // row strip

  __shared__ bf16 sS[4][16 * 32];           // per-wave 16x32 bf16 score tile
  bf16* sp = sS[wave];

  // q fragments are loop-invariant
  v16bf a0 = load_frag(qB, m0, HH_, doff + 0,  lane);
  v16bf a1 = load_frag(qB, m0, HH_, doff + 32, lane);

  v8f acc[4] = {};
  int mofs = (lane >> 4) * 8;
  int ncol = lane & 15;

  for (int j0 = 0; j0 < n; j0 += 32) {
    // stage 1: S[16x32] = q_strip * k_chunk^T  (K = dh = 64)
    v8f s0 = {}, s1 = {};
    v16bf b0 = load_frag(kB, j0,      HH_, doff + 0, lane);
    v16bf b1 = load_frag(kB, j0 + 16, HH_, doff + 0, lane);
    s0 = wmma_bf16(a0, b0, s0);
    s1 = wmma_bf16(a0, b1, s1);
    b0 = load_frag(kB, j0,      HH_, doff + 32, lane);
    b1 = load_frag(kB, j0 + 16, HH_, doff + 32, lane);
    s0 = wmma_bf16(a1, b0, s0);
    s1 = wmma_bf16(a1, b1, s1);

    // branch-free tanh epilogue, convert to bf16, stage into LDS [16][32]
#pragma unroll
    for (int r = 0; r < 8; ++r) {
      sp[(mofs + r) * 32 + ncol]      = (bf16)fast_tanh(s0[r]);
      sp[(mofs + r) * 32 + 16 + ncol] = (bf16)fast_tanh(s1[r]);
    }
    // same-wave LDS ops are in order; reload as an A fragment
    v16bf sa = *(const v16bf*)&sp[(lane & 15) * 32 + ((lane >> 4) << 4)];

    // stage 2: acc[16x64] += S * v_chunk  (B rows = vT[d][j0..])
#pragma unroll
    for (int t = 0; t < 4; ++t) {
      v16bf bv = load_frag(vT, doff + t * 16, n, j0, lane);
      acc[t] = wmma_bf16(sa, bv, acc[t]);
    }
  }
#pragma unroll
  for (int t = 0; t < 4; ++t)
#pragma unroll
    for (int r = 0; r < 8; ++r)
      attB[(size_t)(m0 + mofs + r) * HH_ + doff + t * 16 + ncol] =
          (bf16)acc[t][r];
}

// ---------------------------------------------------------------------------
// Small utility kernels
// ---------------------------------------------------------------------------
__global__ void k_cvt(const float* __restrict__ s, bf16* __restrict__ d, int n) {
  int i = blockIdx.x * blockDim.x + threadIdx.x;
  if (i < n) d[i] = (bf16)s[i];
}

// dst[C x R] (bf16) = transpose of src[R x (>=C)] (f32), with col offset
__global__ void k_transpose(const float* __restrict__ src, int ld, int coloff,
                            int R, int C, bf16* __restrict__ dst) {
  int idx = blockIdx.x * blockDim.x + threadIdx.x;
  if (idx >= R * C) return;
  int r = idx / C, c = idx - r * C;
  dst[(size_t)c * R + r] = (bf16)src[(size_t)r * ld + coloff + c];
}

__global__ __launch_bounds__(256) void k_rowrsqrt(const float* __restrict__ corr,
                                                  float* __restrict__ rn, int n) {
  int row = blockIdx.x;
  const float* p = corr + (size_t)row * n;
  float s = 0.f;
  for (int j = threadIdx.x; j < n; j += 256) { float v = p[j]; s += v * v; }
  __shared__ float red[256];
  red[threadIdx.x] = s; __syncthreads();
  for (int o = 128; o > 0; o >>= 1) {
    if (threadIdx.x < o) red[threadIdx.x] += red[threadIdx.x + o];
    __syncthreads();
  }
  if (threadIdx.x == 0) rn[row] = __builtin_amdgcn_rsqf(red[0] + 1e-30f);
}

// grid = (n/256, n): row = blockIdx.y, col chunked by blockIdx.x (no int div)
__global__ void k_fuse_A(const float* __restrict__ AG,
                         const float* __restrict__ corr,
                         const float* __restrict__ rn,
                         bf16* __restrict__ Abf, int n) {
  int col = blockIdx.x * 256 + threadIdx.x;
  int row = blockIdx.y;
  size_t idx = (size_t)row * n + col;
  Abf[idx] = (bf16)(AG[idx] + corr[idx] * rn[row]);
}

__global__ void k_gate(const float* __restrict__ hw, const float* __restrict__ f,
                       float* __restrict__ x, bf16* __restrict__ xB, int n) {
  int idx = blockIdx.x * blockDim.x + threadIdx.x;
  if (idx >= n * HH_) return;
  int i = idx >> 7, h = idx & 127;
  float f0 = f[(size_t)i * 256 + h];
  float f1 = f[(size_t)i * 256 + 128 + h];
  float h1 = hw[(size_t)i * 384 + 128 + h];
  float h2 = hw[(size_t)i * 384 + 256 + h];
  float v = h2 + fmaxf(f0 + h1, 0.f) * f1;
  x[idx] = v; xB[idx] = (bf16)v;
}

__global__ void k_ffn(float* __restrict__ x, bf16* __restrict__ xB,
                      const float* __restrict__ t, const float* __restrict__ bias,
                      int total) {
  int idx = blockIdx.x * blockDim.x + threadIdx.x;
  if (idx >= total) return;
  int h = idx & 127;
  float v = x[idx] + fmaxf(t[idx] + bias[h], 0.f);
  x[idx] = v; xB[idx] = (bf16)v;
}

// fl[row] and 1/fl[row]
__global__ __launch_bounds__(128) void k_fl(const float* __restrict__ x,
                                            float* __restrict__ fl,
                                            float* __restrict__ invfl) {
  int row = blockIdx.x;
  float v = x[(size_t)row * HH_ + threadIdx.x];
  float s = v * v + 1e-6f;
  __shared__ float red[128];
  red[threadIdx.x] = s; __syncthreads();
  for (int o = 64; o > 0; o >>= 1) {
    if (threadIdx.x < o) red[threadIdx.x] += red[threadIdx.x + o];
    __syncthreads();
  }
  if (threadIdx.x == 0) {
    float r = sqrtf(red[0]);
    fl[row] = r;
    invfl[row] = __builtin_amdgcn_rcpf(r);
  }
}

// FindNeighbors: row softmax stats + top-3 + softmax-of-top3 weighted gather
__global__ __launch_bounds__(256) void k_neighbors(
    const float* __restrict__ sc, const float* __restrict__ invfl,
    const float* __restrict__ x, float* __restrict__ out, int n) {
  int row = blockIdx.x, tid = threadIdx.x;
  const float* srow = sc + (size_t)row * n;
  float inv_i = invfl[row];
  __shared__ float red[256];

  // pass 1: row max of scaled score (all multiplies, no division)
  float mx = -1e30f;
  for (int j = tid; j < n; j += 256) mx = fmaxf(mx, srow[j] * inv_i * invfl[j]);
  red[tid] = mx; __syncthreads();
  for (int o = 128; o > 0; o >>= 1) {
    if (tid < o) red[tid] = fmaxf(red[tid], red[tid + o]);
    __syncthreads();
  }
  float M = red[0]; __syncthreads();

  // pass 2: sum(exp) + per-thread top-3
  float sum = 0.f;
  float tv0 = -1e30f, tv1 = -1e30f, tv2 = -1e30f;
  int   ti0 = 0, ti1 = 0, ti2 = 0;
  for (int j = tid; j < n; j += 256) {
    float s = srow[j] * inv_i * invfl[j];
    sum += __expf(s - M);
    if (s > tv2) {
      if (s > tv0)      { tv2 = tv1; ti2 = ti1; tv1 = tv0; ti1 = ti0; tv0 = s; ti0 = j; }
      else if (s > tv1) { tv2 = tv1; ti2 = ti1; tv1 = s; ti1 = j; }
      else              { tv2 = s; ti2 = j; }
    }
  }
  red[tid] = sum; __syncthreads();
  for (int o = 128; o > 0; o >>= 1) {
    if (tid < o) red[tid] += red[tid + o];
    __syncthreads();
  }
  float invZ = __builtin_amdgcn_rcpf(red[0]);

  __shared__ float cv[256 * 3];
  __shared__ int   ci[256 * 3];
  cv[tid * 3 + 0] = tv0; ci[tid * 3 + 0] = ti0;
  cv[tid * 3 + 1] = tv1; ci[tid * 3 + 1] = ti1;
  cv[tid * 3 + 2] = tv2; ci[tid * 3 + 2] = ti2;
  __syncthreads();

  __shared__ float cw[3];
  __shared__ int   bi[3];
  if (tid == 0) {
    float bv0 = -1e30f, bv1 = -1e30f, bv2 = -1e30f;
    int   b0 = 0, b1 = 0, b2 = 0;
    for (int c = 0; c < 256 * 3; ++c) {
      float s = cv[c]; int j = ci[c];
      if (s > bv2) {
        if (s > bv0)      { bv2 = bv1; b2 = b1; bv1 = bv0; b1 = b0; bv0 = s; b0 = j; }
        else if (s > bv1) { bv2 = bv1; b2 = b1; bv1 = s; b1 = j; }
        else              { bv2 = s; b2 = j; }
      }
    }
    float t0 = __expf(bv0 - M) * invZ;
    float t1 = __expf(bv1 - M) * invZ;
    float t2 = __expf(bv2 - M) * invZ;
    float mt = fmaxf(t0, fmaxf(t1, t2));
    float e0 = __expf(t0 - mt), e1 = __expf(t1 - mt), e2 = __expf(t2 - mt);
    float izz = __builtin_amdgcn_rcpf(e0 + e1 + e2);
    cw[0] = e0 * izz; cw[1] = e1 * izz; cw[2] = e2 * izz;
    bi[0] = b0; bi[1] = b1; bi[2] = b2;
  }
  __syncthreads();
  if (tid < HH_) {
    float acc = cw[0] * x[(size_t)bi[0] * HH_ + tid] +
                cw[1] * x[(size_t)bi[1] * HH_ + tid] +
                cw[2] * x[(size_t)bi[2] * HH_ + tid];
    out[(size_t)row * HH_ + tid] = acc;
  }
}

// ---------------------------------------------------------------------------
// Host orchestration
// ---------------------------------------------------------------------------
extern "C" void kernel_launch(void* const* d_in, const int* in_sizes, int n_in,
                              void* d_out, int out_size, void* d_ws, size_t ws_size,
                              hipStream_t stream) {
  (void)in_sizes; (void)n_in; (void)out_size; (void)ws_size;
  const int N = NN_, H = HH_;
  const size_t NNe = (size_t)N * (size_t)N;

  const float* hidden = (const float*)d_in[0];
  const float* AG     = (const float*)d_in[1];
  const float* w_h    = (const float*)d_in[2];
  const float* w_hf   = (const float*)d_in[3];
  const float* wq     = (const float*)d_in[4];
  const float* wk     = (const float*)d_in[5];
  const float* wv     = (const float*)d_in[6];
  const float* ffn_w  = (const float*)d_in[7];
  const float* ffn_b  = (const float*)d_in[8];
  float* out = (float*)d_out;

  // workspace bump allocator
  char* ws = (char*)d_ws;
  size_t off = 0;
  auto alloc = [&](size_t bytes) -> void* {
    void* p = ws + off;
    off += (bytes + 255) & ~(size_t)255;
    return p;
  };
  float* corrF  = (float*)alloc(NNe * 4);          // also reused as fenzi scores
  bf16*  Abf    = (bf16*)alloc(NNe * 2);
  bf16*  hidB   = (bf16*)alloc((size_t)N * H * 2);
  bf16*  whB    = (bf16*)alloc((size_t)384 * H * 2);
  bf16*  whfB   = (bf16*)alloc((size_t)256 * H * 2);
  bf16*  wqB    = (bf16*)alloc((size_t)2 * H * H * 2);
  bf16*  wkB    = (bf16*)alloc((size_t)2 * H * H * 2);
  bf16*  wvB    = (bf16*)alloc((size_t)2 * H * H * 2);
  bf16*  fwB    = (bf16*)alloc((size_t)2 * H * H * 2);
  float* hwF    = (float*)alloc((size_t)N * 384 * 4);
  bf16*  h0T    = (bf16*)alloc((size_t)H * N * 2);
  bf16*  gB     = (bf16*)alloc((size_t)N * H * 2);
  float* fF     = (float*)alloc((size_t)N * 256 * 4);
  float* xF     = (float*)alloc((size_t)N * H * 4);
  bf16*  xB     = (bf16*)alloc((size_t)N * H * 2);
  bf16*  qB     = (bf16*)alloc((size_t)N * H * 2);
  bf16*  kB     = (bf16*)alloc((size_t)N * H * 2);
  float* vF     = (float*)alloc((size_t)N * H * 4);
  bf16*  vT     = (bf16*)alloc((size_t)H * N * 2);
  bf16*  attB   = (bf16*)alloc((size_t)N * H * 2);
  float* tF     = (float*)alloc((size_t)N * H * 4);
  float* rn     = (float*)alloc((size_t)N * 4);
  float* flv    = (float*)alloc((size_t)N * 4);
  float* iflv   = (float*)alloc((size_t)N * 4);

  auto cvt = [&](const float* s, bf16* d, int n) {
    k_cvt<<<(n + 255) / 256, 256, 0, stream>>>(s, d, n);
  };
  auto gemmF = [&](const bf16* A, const bf16* B, float* C, int M, int Nc, int K,
                   int lda, int ldb, int ldc) {
    dim3 g((Nc + 127) / 128, (M + 63) / 64);
    k_gemm_nt<false><<<g, 128, 0, stream>>>(A, B, (void*)C, M, Nc, K, lda, ldb, ldc);
  };
  auto gemmB = [&](const bf16* A, const bf16* B, bf16* C, int M, int Nc, int K,
                   int lda, int ldb, int ldc) {
    dim3 g((Nc + 127) / 128, (M + 63) / 64);
    k_gemm_nt<true><<<g, 128, 0, stream>>>(A, B, (void*)C, M, Nc, K, lda, ldb, ldc);
  };

  // ---- bf16 conversions of all GEMM operands ----
  cvt(hidden, hidB, N * H);
  cvt(w_h,  whB,  384 * H);
  cvt(w_hf, whfB, 256 * H);
  cvt(wq, wqB, 2 * H * H);
  cvt(wk, wkB, 2 * H * H);
  cvt(wv, wvB, 2 * H * H);
  cvt(ffn_w, fwB, 2 * H * H);

  // ---- corr = hidden hidden^T ; A = A_Global + corr/||corr_row|| ----
  gemmF(hidB, hidB, corrF, N, N, H, H, H, N);
  k_rowrsqrt<<<N, 256, 0, stream>>>(corrF, rn, N);
  {
    dim3 g(N / 256, N);
    k_fuse_A<<<g, 256, 0, stream>>>(AG, corrF, rn, Abf, N);
  }

  // ---- GateCell: hw = hidden w_h^T ; f = (A h0) w_hf^T ; gate ----
  gemmF(hidB, whB, hwF, N, 384, H, H, H, 384);
  k_transpose<<<(N * H + 255) / 256, 256, 0, stream>>>(hwF, 384, 0, N, H, h0T);
  gemmB(Abf, h0T, gB, N, H, N, N, N, H);          // g = A @ h0  (B rows = h0^T)
  gemmF(gB, whfB, fF, N, 256, H, H, H, 256);
  k_gate<<<(N * H + 255) / 256, 256, 0, stream>>>(hwF, fF, xF, xB, N);

  // ---- 2 attention blocks ----
  for (int b = 0; b < 2; ++b) {
    const bf16* wqb = wqB + (size_t)b * H * H;
    const bf16* wkb = wkB + (size_t)b * H * H;
    const bf16* wvb = wvB + (size_t)b * H * H;
    const bf16* fwb = fwB + (size_t)b * H * H;
    gemmB(xB, wqb, qB, N, H, H, H, H, H);
    gemmB(xB, wkb, kB, N, H, H, H, H, H);
    gemmF(xB, wvb, vF, N, H, H, H, H, H);
    k_transpose<<<(N * H + 255) / 256, 256, 0, stream>>>(vF, H, 0, N, H, vT);
    {
      dim3 g(N / 64, 2);  // (row tiles, heads)
      k_att<<<g, 128, 0, stream>>>(qB, kB, vT, attB, N);
    }
    gemmF(attB, fwb, tF, N, H, H, H, H, H);
    k_ffn<<<(N * H + 255) / 256, 256, 0, stream>>>(xF, xB, tF, ffn_b + (size_t)b * H, N * H);
  }

  // ---- FindNeighbors ----
  k_fl<<<N, 128, 0, stream>>>(xF, flv, iflv);
  gemmF(xB, xB, corrF, N, N, H, H, H, N);  // fenzi, reusing corr buffer
  k_neighbors<<<N, 256, 0, stream>>>(corrF, iflv, xF, out, N);
}
